// LearnableMahalanobisTopK_23021024707055
// MI455X (gfx1250) — matrix operand order repaired and checked by reading, not compiled, not run on hardware
//
#include <hip/hip_runtime.h>
#include <hip/hip_bf16.h>

// ---------------------------------------------------------------------------
// LearnableMahalanobisTopK for MI455X (gfx1250, wave32, WMMA)
//   Phase 1: x_proj = x @ W^T  via v_wmma_f32_16x16x32_f16 (f16 in, f32 acc)
//   Phase 2: per-node (deg=16) Mahalanobis scores, softmax, stable top-8,
//            renormalize, emit (row, col, weight) sorted by edge index.
// ---------------------------------------------------------------------------

typedef __attribute__((ext_vector_type(16))) _Float16 v16h;
typedef __attribute__((ext_vector_type(8)))  _Float16 h8;
typedef __attribute__((ext_vector_type(4)))  _Float16 h4;
typedef __attribute__((ext_vector_type(8)))  float    v8f;
typedef __attribute__((ext_vector_type(4)))  float    v4f;

#define DIM    128
#define DEG    16
#define TOPK   8
#define EPSF   1e-8f
#define WPITCH 136   // halfs per LDS row of W: 272 B = 16-byte aligned, bank shift 17 (coprime 64)

// ---------------------------------------------------------------------------
// Kernel 1: x_proj[n, j] = sum_k x[n,k] * W[j,k]
// Block = 256 threads = 8 waves; each wave owns a 16-row strip (128 rows/block).
// ---------------------------------------------------------------------------
__global__ __launch_bounds__(256) void gemm_xwT_wmma(const float* __restrict__ x,
                                                     const float* __restrict__ W,
                                                     float* __restrict__ xp,
                                                     int n) {
  __shared__ __attribute__((aligned(16))) _Float16 Wh[DIM * WPITCH];

  const int t = threadIdx.x;

  // Stage W (128x128 f32, row-major) into LDS as f16 with padded pitch.
  // 4096 float4 elements, 16 per thread, fully coalesced.
  for (int i = t; i < (DIM * DIM) / 4; i += 256) {
    const int r  = i >> 5;       // 32 float4 per row
    const int c4 = i & 31;
    v4f w4 = *(const v4f*)(W + r * DIM + c4 * 4);
    h4 hv;
    hv[0] = (_Float16)w4[0]; hv[1] = (_Float16)w4[1];
    hv[2] = (_Float16)w4[2]; hv[3] = (_Float16)w4[3];
    *(h4*)(&Wh[r * WPITCH + c4 * 4]) = hv;
  }
  __syncthreads();

  const int lane = t & 31;
  const int wave = t >> 5;
  const int m    = lane & 15;    // A: row M   | B/C: col N
  const int kb   = lane >> 4;    // half-wave selects K sub-block

  const int rowBase = blockIdx.x * 128 + wave * 16;
  if (rowBase >= n) return;                // whole strip out of range (uniform per wave)

  const bool fullStrip = (rowBase + 16 <= n);

  int arow = rowBase + m;
  if (arow >= n) arow = n - 1;             // clamp: keep EXEC all-1s for WMMA
  const float* xr = x + (size_t)arow * DIM;

  __builtin_prefetch(xr, 0, 1);            // global_prefetch_b8

  // Build 4 A-fragments (K-steps of 32) per the ISA 16-bit A layout:
  // lane (m, kb); elems 0..7 -> K = K0 + kb*8 + {0..7}; elems 8..15 -> K0+16+kb*8+{0..7}
  v16h A[4];
#pragma unroll
  for (int s = 0; s < 4; ++s) {
    const int K0 = s * 32;
    v4f c0 = *(const v4f*)(xr + K0 + kb * 8);
    v4f c1 = *(const v4f*)(xr + K0 + kb * 8 + 4);
    v4f c2 = *(const v4f*)(xr + K0 + 16 + kb * 8);
    v4f c3 = *(const v4f*)(xr + K0 + 16 + kb * 8 + 4);
    v16h a;
#pragma unroll
    for (int i = 0; i < 4; ++i) {
      a[i]      = (_Float16)c0[i];
      a[4 + i]  = (_Float16)c1[i];
      a[8 + i]  = (_Float16)c2[i];
      a[12 + i] = (_Float16)c3[i];
    }
    A[s] = a;
  }

  // Base output pointer for this lane's column within this wave's strip:
  // C/D layout: VGPR r -> M = r + 8*kb, lane&15 -> N.
  float* outBase = xp + (size_t)(rowBase + 8 * kb) * DIM + m;

  // 8 output column tiles of 16; 4 WMMAs (K=32 each) per tile.
#pragma unroll
  for (int tile = 0; tile < 8; ++tile) {
    v8f acc = {};
    const int bn = tile * 16 + m;          // B column N (== W row index)
#pragma unroll
    for (int s = 0; s < 4; ++s) {
      // B layout: elems 0..7 -> K = kb*8 + {0..7}; elems 8..15 -> K = 16 + kb*8 + {0..7}.
      const _Float16* bp = &Wh[bn * WPITCH + s * 32 + kb * 8];
      h8 blo = *(const h8*)bp;
      h8 bhi = *(const h8*)(bp + 16);
      v16h b;
#pragma unroll
      for (int i = 0; i < 8; ++i) { b[i] = blo[i]; b[8 + i] = bhi[i]; }
      acc = __builtin_amdgcn_wmma_f32_16x16x32_f16(
          /*neg_a=*/false, A[s], /*neg_b=*/false, b,
          /*c_mod=*/(short)0, acc, /*reuse_a=*/false, /*reuse_b=*/false);
    }

    if (fullStrip) {
      // Fast path (390/391 blocks): plain stores, immediate offsets r*512B.
      float* op = outBase + tile * 16;
#pragma unroll
      for (int r = 0; r < 8; ++r) op[(size_t)r * DIM] = acc[r];
    } else {
      // Boundary strip: per-row guard.
#pragma unroll
      for (int r = 0; r < 8; ++r) {
        const int row = rowBase + r + 8 * kb;
        if (row < n) xp[(size_t)row * DIM + tile * 16 + m] = acc[r];
      }
    }
  }
}

// ---------------------------------------------------------------------------
// Kernel 2: per-node scoring + softmax + stable top-8 + renorm + emit.
// One wave per node (low 16 lanes = 16 edges); 8 nodes per 256-thread block.
// x_proj gathers hit the 192 MB L2 (x_proj is 25.6 MB).
// ---------------------------------------------------------------------------
__global__ __launch_bounds__(256) void edge_softmax_topk(const float* __restrict__ xp,
                                                         const int* __restrict__ ei,
                                                         const float* __restrict__ log_diag,
                                                         const float* __restrict__ log_temp,
                                                         int* __restrict__ out_row,
                                                         int* __restrict__ out_col,
                                                         float* __restrict__ out_w,
                                                         int n) {
  __shared__ __attribute__((aligned(16))) float sdiag[DIM];
  __shared__ __attribute__((aligned(16))) float sxp[8][DIM];

  const int t    = threadIdx.x;
  const int lane = t & 31;
  const int wave = t >> 5;

  if (t < DIM) {
    const float v  = log_diag[t];
    const float sp = (v > 20.0f) ? v : log1pf(__expf(v));   // softplus
    sdiag[t] = sp + EPSF;
  }

  const int node = blockIdx.x * 8 + wave;
  const int nn   = node < n ? node : n - 1;
  // Each wave stages its own node's projected row (32 lanes x float4).
  *(v4f*)(&sxp[wave][lane * 4]) = *(const v4f*)(xp + (size_t)nn * DIM + lane * 4);
  __syncthreads();

  const float invtemp = 1.0f / __expf(log_temp[0]);

  if (lane < DEG && node < n) {
    const int E   = n * DEG;
    const int col = ei[E + node * DEG + lane];
    const float* xc = xp + (size_t)col * DIM;

    float acc = 0.0f;
#pragma unroll 4
    for (int d0 = 0; d0 < DIM; d0 += 4) {
      v4f c = *(const v4f*)(xc + d0);
#pragma unroll
      for (int j = 0; j < 4; ++j) {
        const float diff = sxp[wave][d0 + j] - c[j];
        acc = fmaf(sdiag[d0 + j] * diff, diff, acc);
      }
    }
    const float s = -acc * invtemp;

    // softmax over the 16 lanes (butterfly, width 16)
    float mx = s;
    for (int off = 8; off; off >>= 1) mx = fmaxf(mx, __shfl_xor(mx, off, 16));
    const float ex = __expf(s - mx);
    float sum = ex;
    for (int off = 8; off; off >>= 1) sum += __shfl_xor(sum, off, 16);
    const float base = ex / sum;            // exp(s - lse)

    // stable top-k rank (ties -> lower edge index wins, matches lax.top_k)
    int rank = 0;
#pragma unroll
    for (int j = 0; j < DEG; ++j) {
      const float sj = __shfl(s, j, 16);
      rank += (sj > s) || (sj == s && j < lane);
    }
    const bool selected = rank < TOPK;

    float selv = selected ? base : 0.0f;
    float rs = selv;
    for (int off = 8; off; off >>= 1) rs += __shfl_xor(rs, off, 16);
    rs = fmaxf(rs, EPSF);

    // output slot = position among selected edges in ascending edge-index order
    const unsigned mask = (unsigned)__ballot(selected);
    const int pos = __popc(mask & ((1u << lane) - 1u));

    if (selected) {
      const int o = node * TOPK + pos;
      out_row[o] = node;
      out_col[o] = col;
      out_w[o]   = selv / rs;               // == w_st at selected positions
    }
  }
}

// ---------------------------------------------------------------------------
extern "C" void kernel_launch(void* const* d_in, const int* in_sizes, int n_in,
                              void* d_out, int out_size, void* d_ws, size_t ws_size,
                              hipStream_t stream) {
  const float* x        = (const float*)d_in[0];   // (N, 128) f32
  const int*   ei       = (const int*)d_in[1];     // (2, N*16) i32
  const float* W        = (const float*)d_in[2];   // (128, 128) f32
  const float* log_diag = (const float*)d_in[3];   // (128,) f32
  const float* log_temp = (const float*)d_in[4];   // scalar f32
  // d_in[5] = topk (compile-time TOPK)

  const int n = in_sizes[0] / DIM;                 // 50000

  float* xp      = (float*)d_ws;                   // N*128 f32 scratch (25.6 MB)
  int*   out_row = (int*)d_out;                    // pruned_edge_index[0]
  int*   out_col = out_row + n * TOPK;             // pruned_edge_index[1]
  float* out_w   = (float*)d_out + 2 * n * TOPK;   // pruned_edge_weight

  const int gblocks = (n + 127) / 128;
  gemm_xwT_wmma<<<gblocks, 256, 0, stream>>>(x, W, xp, n);

  const int eblocks = (n + 7) / 8;
  edge_softmax_topk<<<eblocks, 256, 0, stream>>>(xp, ei, log_diag, log_temp,
                                                 out_row, out_col, out_w, n);
}
